// TransformerBlock_5136780886148
// MI455X (gfx1250) — matrix operand order repaired and checked by reading, not compiled, not run on hardware
//
#include <hip/hip_runtime.h>
#include <hip/hip_bf16.h>

typedef __attribute__((ext_vector_type(16))) _Float16 v16h;
typedef __attribute__((ext_vector_type(8)))  _Float16 v8h;
typedef __attribute__((ext_vector_type(8)))  float    v8f;
typedef __attribute__((ext_vector_type(4)))  int      v4i;

#define GLOBAL_AS __attribute__((address_space(1)))
#define LDS_AS    __attribute__((address_space(3)))

#if defined(__has_builtin)
#if __has_builtin(__builtin_amdgcn_global_load_async_to_lds_b128)
#define HAVE_ASYNC_LDS 1
#endif
#endif

static constexpr int Dn  = 1024;
static constexpr int Sn  = 1024;
static constexpr int Bn  = 32;
static constexpr int TPB = 256;

// ---------------------------------------------------------------------------
// WMMA fragment helpers (wave32, 16x16x32 f16 -> f32)
// A-matrix 16x32 (M x K): lane m (0..15) holds K=kb..kb+7 in VGPR0-3 and
// K=kb+16..kb+23 in VGPR4-7, kb = (lane>=16)*8.   (ISA 7.12.2)
// ---------------------------------------------------------------------------
__device__ __forceinline__ v16h load_fragA(const _Float16* base, int ld) {
  const int lane = threadIdx.x & 31;
  const int r    = lane & 15;
  const int kb   = (lane >> 4) << 3;     // 0 or 8
  const _Float16* p = base + r * ld + kb;
  v8h lo = *(const v8h*)(p);             // K = kb .. kb+7      -> VGPR0-3
  v8h hi = *(const v8h*)(p + 16);        // K = kb+16 .. kb+23  -> VGPR4-7
  v16h f;
#pragma unroll
  for (int i = 0; i < 8; i++) { f[i] = lo[i]; f[i + 8] = hi[i]; }
  return f;
}

// B-matrix 32x16 (K x N): lane n (0..15) holds K=0..7 and K=16..23,
// lane n+16 holds K=8..15 and K=24..31.  Source rows indexed by N, K contiguous.
__device__ __forceinline__ v16h load_fragB(const _Float16* base, int ld) {
  const int lane = threadIdx.x & 31;
  const int r    = lane & 15;            // N index
  const int kb   = (lane >> 4) << 3;     // 0 or 8
  const _Float16* p = base + r * ld + kb;
  v8h lo = *(const v8h*)(p);
  v8h hi = *(const v8h*)(p + 16);
  v16h f;
#pragma unroll
  for (int i = 0; i < 8; i++) { f[i] = lo[i]; f[i + 8] = hi[i]; }
  return f;
}

#define WMMA_F16(a, b, c) \
  __builtin_amdgcn_wmma_f32_16x16x32_f16(false, (a), false, (b), (short)0, (c), false, false)

// ---------------------------------------------------------------------------
// fp32 -> f16 weight conversion
// ---------------------------------------------------------------------------
__global__ __launch_bounds__(TPB) void cvt_f32_f16(const float* __restrict__ src,
                                                   _Float16* __restrict__ dst, int n) {
  int i = blockIdx.x * blockDim.x + threadIdx.x;
  if (i < n) dst[i] = (_Float16)src[i];
}

// ---------------------------------------------------------------------------
// Attention: one block = (batch b, 16-row Q tile).  8 waves x 128 K-cols.
// ---------------------------------------------------------------------------
__global__ __launch_bounds__(TPB) void attn_kernel(
    const float* __restrict__ Q, const float* __restrict__ K, const float* __restrict__ V,
    const float* __restrict__ gamma, const float* __restrict__ beta,
    float* __restrict__ X32, _Float16* __restrict__ Xh) {
  extern __shared__ __align__(16) char smem[];
  _Float16* Qs   = (_Float16*)smem;               // 16x1024 f16   (32 KB)
  float*    Sf   = (float*)(smem + 32768);        // 16x1024 f32   (64 KB)
  _Float16* Vt   = (_Float16*)(smem + 32768);     // alias: 1024x32 f16
  _Float16* Ps   = (_Float16*)(smem + 98304);     // 16x1024 f16   (32 KB)
  _Float16* Ks   = (_Float16*)(smem + 131072);    // 8 waves x 16x32 (8 KB)
  float*    red0 = (float*)(smem + 139264);       // 16x16
  float*    red1 = red0 + 256;

  const int tid  = threadIdx.x;
  const int wave = tid >> 5;
  const int lane = tid & 31;
  const int b    = blockIdx.x >> 6;   // / (Sn/16)
  const int qt   = blockIdx.x & 63;

  const float* Qb = Q + ((size_t)b * Sn + qt * 16) * Dn;
  const float* Kb = K + (size_t)b * Sn * Dn;
  const float* Vb = V + (size_t)b * Sn * Dn;

  // ---- Phase A: stage Q tile (16x1024) as f16 into LDS ----
  for (int i = tid; i < 16 * Dn / 4; i += TPB) {
    float4 q4 = ((const float4*)Qb)[i];
    int o = i * 4;
    Qs[o + 0] = (_Float16)q4.x; Qs[o + 1] = (_Float16)q4.y;
    Qs[o + 2] = (_Float16)q4.z; Qs[o + 3] = (_Float16)q4.w;
  }
  __syncthreads();

  // ---- Phase B: scores S = (Q K^T)/scale; wave owns K rows [wave*128, +128) ----
  const int kr0 = wave * 128;
  _Float16* Ksw = Ks + wave * 512;
  v8f zero = {0.f, 0.f, 0.f, 0.f, 0.f, 0.f, 0.f, 0.f};
  v8f accS[8];
#pragma unroll
  for (int t = 0; t < 8; t++) accS[t] = zero;

  for (int dc = 0; dc < Dn / 32; dc++) {
    v16h a = load_fragA(Qs + dc * 32, Dn);
    if (dc + 1 < Dn / 32)
      __builtin_prefetch(Kb + (size_t)kr0 * Dn + (dc + 1) * 32, 0, 1);
#pragma unroll
    for (int t = 0; t < 8; t++) {
      // stage K[kr0+t*16 .. +16)[dc*32 .. +32) as f16 (wave-local, DS in-order)
#pragma unroll
      for (int j = 0; j < 4; j++) {
        int idx = lane + j * 32;               // 0..127 float4 slots
        int r   = idx >> 3;
        int c4  = (idx & 7) << 2;
        float4 k4 = *(const float4*)(Kb + (size_t)(kr0 + t * 16 + r) * Dn + dc * 32 + c4);
        _Float16* d = Ksw + r * 32 + c4;
        d[0] = (_Float16)k4.x; d[1] = (_Float16)k4.y;
        d[2] = (_Float16)k4.z; d[3] = (_Float16)k4.w;
      }
      v16h bf = load_fragB(Ksw, 32);
      accS[t] = WMMA_F16(a, bf, accS[t]);
    }
  }
  {
    const float inv_scale = 1.0f / (32.0f + 1e-8f);   // sqrt(1024)+eps
    const int n = lane & 15, rb = (lane >> 4) * 8;
#pragma unroll
    for (int t = 0; t < 8; t++) {
      float* basep = Sf + kr0 + t * 16;
#pragma unroll
      for (int v = 0; v < 8; v++) basep[(rb + v) * Dn + n] = accS[t][v] * inv_scale;
    }
  }
  __syncthreads();

  // ---- Phase C: softmax over each of 16 rows (16 threads per row) ----
  const int row = tid >> 4, sub = tid & 15;
  {
    float mx = -3.0e38f;
    for (int c = sub * 64; c < sub * 64 + 64; c++) mx = fmaxf(mx, Sf[row * Dn + c]);
    red0[row * 16 + sub] = mx;
    __syncthreads();
    float rmax = red0[row * 16];
#pragma unroll
    for (int i = 1; i < 16; i++) rmax = fmaxf(rmax, red0[row * 16 + i]);
    float s = 0.f;
    for (int c = sub * 64; c < sub * 64 + 64; c++) {
      float e = __expf(Sf[row * Dn + c] - rmax);
      Ps[row * Dn + c] = (_Float16)e;
      s += e;
    }
    red1[row * 16 + sub] = s;
    __syncthreads();
    float rsum = 0.f;
#pragma unroll
    for (int i = 0; i < 16; i++) rsum += red1[row * 16 + i];
    float rinv = 1.0f / rsum;
    for (int c = sub * 64; c < sub * 64 + 64; c++)
      Ps[row * Dn + c] = (_Float16)((float)Ps[row * Dn + c] * rinv);
  }

  // ---- Phase D: O = P @ V; wave owns output cols [wave*128, +128) ----
  v8f accO[8];
#pragma unroll
  for (int t = 0; t < 8; t++) accO[t] = zero;

  for (int kc = 0; kc < Sn / 32; kc++) {
    __syncthreads();                       // WAR on Vt (and first iter: Sf readers done)
#pragma unroll 4
    for (int i = 0; i < 32; i++) {         // stage V chunk transposed: Vt[col][k]
      float4 v4 = *(const float4*)(Vb + (size_t)(kc * 32 + i) * Dn + tid * 4);
      int c0 = tid * 4;
      Vt[(c0 + 0) * 32 + i] = (_Float16)v4.x;
      Vt[(c0 + 1) * 32 + i] = (_Float16)v4.y;
      Vt[(c0 + 2) * 32 + i] = (_Float16)v4.z;
      Vt[(c0 + 3) * 32 + i] = (_Float16)v4.w;
    }
    __syncthreads();
    v16h a = load_fragA(Ps + kc * 32, Dn);
#pragma unroll
    for (int t = 0; t < 8; t++) {
      v16h bf = load_fragB(Vt + (wave * 128 + t * 16) * 32, 32);
      accO[t] = WMMA_F16(a, bf, accO[t]);
    }
  }
  __syncthreads();

  // ---- Phase E: Y = Q + O; LayerNorm -> X32 (f32) and Xh (f16) ----
  {
    const int n = lane & 15, rb = (lane >> 4) * 8;
#pragma unroll
    for (int t = 0; t < 8; t++) {
      float* basep = Sf + wave * 128 + t * 16;
#pragma unroll
      for (int v = 0; v < 8; v++) basep[(rb + v) * Dn + n] = accO[t][v];
    }
  }
  __syncthreads();
  {
    float psum = 0.f, psq = 0.f;
    for (int c = sub * 64; c < sub * 64 + 64; c++) {
      float y = Qb[row * Dn + c] + Sf[row * Dn + c];
      Sf[row * Dn + c] = y;
      psum += y; psq += y * y;
    }
    red0[row * 16 + sub] = psum;
    red1[row * 16 + sub] = psq;
    __syncthreads();
    float s1 = 0.f, s2 = 0.f;
#pragma unroll
    for (int i = 0; i < 16; i++) { s1 += red0[row * 16 + i]; s2 += red1[row * 16 + i]; }
    float mu   = s1 * (1.0f / Dn);
    float var  = s2 * (1.0f / Dn) - mu * mu;
    float rstd = rsqrtf(var + 1e-5f);
    size_t gro = ((size_t)b * Sn + qt * 16 + row) * Dn;
    for (int c = sub * 64; c < sub * 64 + 64; c++) {
      float x = (Sf[row * Dn + c] - mu) * rstd * gamma[c] + beta[c];
      X32[gro + c] = x;
      Xh[gro + c]  = (_Float16)x;
    }
  }
}

// ---------------------------------------------------------------------------
// Shared helper for FFN weight staging: copy W[n=0..1023][dc*32..+32) -> Ws,
// using async global->LDS DMA when the toolchain exposes it.
// The builtin takes pointers to int-vec4 (b128) in AS1 / AS3.
// ---------------------------------------------------------------------------
__device__ __forceinline__ void stage_w_tile(const _Float16* __restrict__ W,
                                             _Float16* __restrict__ Ws,
                                             int dc, int tid) {
  for (int i = tid; i < Dn * 32 / 8; i += TPB) {
    int r = i >> 2, c8 = (i & 3) * 8;
    const _Float16* g = W + (size_t)r * Dn + dc * 32 + c8;
    _Float16*       l = Ws + r * 32 + c8;
#ifdef HAVE_ASYNC_LDS
    __builtin_amdgcn_global_load_async_to_lds_b128(
        (GLOBAL_AS v4i*)g, (LDS_AS v4i*)l, 0, 0);
#else
    *(v8h*)l = *(const v8h*)g;
#endif
  }
#ifdef HAVE_ASYNC_LDS
#if __has_builtin(__builtin_amdgcn_s_wait_asynccnt)
  __builtin_amdgcn_s_wait_asynccnt(0);
#else
  asm volatile("s_wait_asynccnt 0x0" ::: "memory");
#endif
#endif
}

// ---------------------------------------------------------------------------
// FFN1: H = relu(X @ W1^T + b1)  (f16 out).  Block = 16 rows x 1024 cols.
// ---------------------------------------------------------------------------
__global__ __launch_bounds__(TPB) void ffn1_kernel(
    const _Float16* __restrict__ Xh, const _Float16* __restrict__ W1h,
    const float* __restrict__ b1, _Float16* __restrict__ Hh) {
  extern __shared__ __align__(16) char smem[];
  _Float16* As = (_Float16*)smem;            // 16x32 f16 (1 KB)
  _Float16* Ws = (_Float16*)(smem + 1024);   // 1024x32 f16 (64 KB)

  const int tid = threadIdx.x, wave = tid >> 5, lane = tid & 31;
  const int mt  = blockIdx.x;
  const _Float16* Xrow = Xh + (size_t)mt * 16 * Dn;

  v8f zero = {0.f, 0.f, 0.f, 0.f, 0.f, 0.f, 0.f, 0.f};
  v8f acc[8];
#pragma unroll
  for (int t = 0; t < 8; t++) acc[t] = zero;

  for (int dc = 0; dc < Dn / 32; dc++) {
    __syncthreads();
    if (tid < 64) {
      int r = tid >> 2, c8 = (tid & 3) * 8;
      *(v8h*)(As + r * 32 + c8) = *(const v8h*)(Xrow + (size_t)r * Dn + dc * 32 + c8);
    }
    stage_w_tile(W1h, Ws, dc, tid);
    __syncthreads();
    v16h a = load_fragA(As, 32);
#pragma unroll
    for (int t = 0; t < 8; t++) {
      v16h bf = load_fragB(Ws + (wave * 128 + t * 16) * 32, 32);
      acc[t] = WMMA_F16(a, bf, acc[t]);
    }
  }
  const int n = lane & 15, rb = (lane >> 4) * 8;
#pragma unroll
  for (int t = 0; t < 8; t++) {
    int n0 = wave * 128 + t * 16;
    float bias = b1[n0 + n];
#pragma unroll
    for (int v = 0; v < 8; v++) {
      float h = fmaxf(acc[t][v] + bias, 0.0f);
      Hh[((size_t)mt * 16 + rb + v) * Dn + n0 + n] = (_Float16)h;
    }
  }
}

// ---------------------------------------------------------------------------
// FFN2: out = LayerNorm(H @ W2^T + b2 + X)
// ---------------------------------------------------------------------------
__global__ __launch_bounds__(TPB) void ffn2_kernel(
    const _Float16* __restrict__ Hh, const _Float16* __restrict__ W2h,
    const float* __restrict__ b2, const float* __restrict__ X32,
    const float* __restrict__ gamma, const float* __restrict__ beta,
    float* __restrict__ out) {
  extern __shared__ __align__(16) char smem[];
  _Float16* As   = (_Float16*)smem;             // 1 KB
  _Float16* Ws   = (_Float16*)(smem + 1024);    // 64 KB
  float*    Yf   = (float*)(smem + 66560);      // 16x1024 f32 (64 KB)
  float*    red0 = (float*)(smem + 132096);
  float*    red1 = red0 + 256;

  const int tid = threadIdx.x, wave = tid >> 5, lane = tid & 31;
  const int mt  = blockIdx.x;
  const _Float16* Hrow = Hh + (size_t)mt * 16 * Dn;

  v8f zero = {0.f, 0.f, 0.f, 0.f, 0.f, 0.f, 0.f, 0.f};
  v8f acc[8];
#pragma unroll
  for (int t = 0; t < 8; t++) acc[t] = zero;

  for (int dc = 0; dc < Dn / 32; dc++) {
    __syncthreads();
    if (tid < 64) {
      int r = tid >> 2, c8 = (tid & 3) * 8;
      *(v8h*)(As + r * 32 + c8) = *(const v8h*)(Hrow + (size_t)r * Dn + dc * 32 + c8);
    }
    stage_w_tile(W2h, Ws, dc, tid);
    __syncthreads();
    v16h a = load_fragA(As, 32);
#pragma unroll
    for (int t = 0; t < 8; t++) {
      v16h bf = load_fragB(Ws + (wave * 128 + t * 16) * 32, 32);
      acc[t] = WMMA_F16(a, bf, acc[t]);
    }
  }
  // epilogue: bias into LDS, residual, LayerNorm
  {
    const int n = lane & 15, rb = (lane >> 4) * 8;
#pragma unroll
    for (int t = 0; t < 8; t++) {
      int n0 = wave * 128 + t * 16;
      float bias = b2[n0 + n];
#pragma unroll
      for (int v = 0; v < 8; v++)
        Yf[(rb + v) * Dn + n0 + n] = acc[t][v] + bias;
    }
  }
  __syncthreads();
  {
    const int row = tid >> 4, sub = tid & 15;
    size_t gro = ((size_t)mt * 16 + row) * Dn;
    float psum = 0.f, psq = 0.f;
    for (int c = sub * 64; c < sub * 64 + 64; c++) {
      float y = Yf[row * Dn + c] + X32[gro + c];
      Yf[row * Dn + c] = y;
      psum += y; psq += y * y;
    }
    red0[row * 16 + sub] = psum;
    red1[row * 16 + sub] = psq;
    __syncthreads();
    float s1 = 0.f, s2 = 0.f;
#pragma unroll
    for (int i = 0; i < 16; i++) { s1 += red0[row * 16 + i]; s2 += red1[row * 16 + i]; }
    float mu   = s1 * (1.0f / Dn);
    float var  = s2 * (1.0f / Dn) - mu * mu;
    float rstd = rsqrtf(var + 1e-5f);
    for (int c = sub * 64; c < sub * 64 + 64; c++)
      out[gro + c] = (Yf[row * Dn + c] - mu) * rstd * gamma[c] + beta[c];
  }
}

// ---------------------------------------------------------------------------
// Host launch
// ---------------------------------------------------------------------------
extern "C" void kernel_launch(void* const* d_in, const int* in_sizes, int n_in,
                              void* d_out, int out_size, void* d_ws, size_t ws_size,
                              hipStream_t stream) {
  const float* Q     = (const float*)d_in[0];
  const float* K     = (const float*)d_in[1];
  const float* V     = (const float*)d_in[2];
  const float* W1    = (const float*)d_in[3];
  const float* b1    = (const float*)d_in[4];
  const float* W2    = (const float*)d_in[5];
  const float* b2    = (const float*)d_in[6];
  const float* gamma = (const float*)d_in[7];
  const float* beta  = (const float*)d_in[8];
  float* out = (float*)d_out;

  char* ws = (char*)d_ws;
  _Float16* Xh  = (_Float16*)(ws);                          // 64 MB
  _Float16* Hh  = (_Float16*)(ws + ((size_t)64 << 20));     // 64 MB
  float*    X32 = (float*)(ws + ((size_t)128 << 20));       // 128 MB
  _Float16* W1h = (_Float16*)(ws + ((size_t)256 << 20));    // 2 MB
  _Float16* W2h = (_Float16*)(ws + ((size_t)258 << 20));    // 2 MB

  const int nW = Dn * Dn;
  cvt_f32_f16<<<nW / TPB, TPB, 0, stream>>>(W1, W1h, nW);
  cvt_f32_f16<<<nW / TPB, TPB, 0, stream>>>(W2, W2h, nW);

  const int ATTN_SMEM = 141312;   // Qs32K + Sf/Vt64K + Ps32K + Ks8K + red2K
  const int FFN1_SMEM = 66560;    // As1K + Ws64K
  const int FFN2_SMEM = 134144;   // As1K + Ws64K + Yf64K + red2K

  attn_kernel<<<Bn * (Sn / 16), TPB, ATTN_SMEM, stream>>>(Q, K, V, gamma, beta, X32, Xh);
  ffn1_kernel<<<(Bn * Sn) / 16, TPB, FFN1_SMEM, stream>>>(Xh, W1h, b1, Hh);
  ffn2_kernel<<<(Bn * Sn) / 16, TPB, FFN2_SMEM, stream>>>(Hh, W2h, b2, X32, gamma, beta, out);
}
